// GATLayer_70007966925419
// MI455X (gfx1250) — matrix operand order repaired and checked by reading, not compiled, not run on hardware
//
#include <hip/hip_runtime.h>
#include <hip/hip_bf16.h>

// GAT layer, fused for MI455X (gfx1250, wave32).
//   Phase 1: Wh = X@W (f32); WhT = f16 K-major transpose, padded to 80 rows
//            (row 64 = ones -> softmax denominator computed by WMMA itself).
//   Phase 2: fs/fd = Wh@attn halves; Fmax = max(fd).
//   Phase 3: single pass over the 268MB adjacency matrix (HBM roofline floor):
//            branchless in-register probability tile (f16 WMMA A-fragment),
//            5x v_wmma_f32_16x16x32_f16 per 32-col chunk (4 output tiles +
//            denominator tile). fd staged into LDS via the Tensor Data Mover.

#define NN 8192
#define IN_DIM 256
#define OUT_DIM 64
#define LOG2E 1.44269504088896340736f

typedef __attribute__((ext_vector_type(16))) _Float16 v16h;
typedef __attribute__((ext_vector_type(8)))  float    v8f;
typedef unsigned int u32x4 __attribute__((ext_vector_type(4)));
typedef int          i32x4 __attribute__((ext_vector_type(4)));
typedef int          i32x8 __attribute__((ext_vector_type(8)));

// ---------------- Phase 1: Wh = X @ W, plus f16 K-major transpose ----------
__global__ void wh_kernel(const float* __restrict__ X, const float* __restrict__ W,
                          float* __restrict__ Wh, _Float16* __restrict__ WhT) {
    const int tid = threadIdx.x;
    const int col = tid & 63;
    const int row = blockIdx.x * 4 + (tid >> 6);
    const float* xr = X + (size_t)row * IN_DIM;
    float acc = 0.f;
#pragma unroll 8
    for (int k = 0; k < IN_DIM; ++k)
        acc = fmaf(xr[k], W[k * OUT_DIM + col], acc);
    Wh[(size_t)row * OUT_DIM + col] = acc;
    WhT[(size_t)col * NN + row] = (_Float16)acc;   // [col][row] K-major
}

// Pad WhT rows 64..79: row 64 = 1.0 (denominator column), rows 65..79 = 0.
__global__ void whtpad_kernel(_Float16* __restrict__ WhT) {
    const int idx = blockIdx.x * 256 + threadIdx.x;   // 0 .. 16*NN-1
    const int rr  = idx >> 13;                        // 0..15
    WhT[(size_t)(OUT_DIM + rr) * NN + (idx & (NN - 1))] =
        (rr == 0) ? (_Float16)1.0f : (_Float16)0.0f;
}

// ---------------- Phase 2a: fs = Wh@a[:64], fd = Wh@a[64:] -----------------
__global__ void f_kernel(const float* __restrict__ Wh, const float* __restrict__ attn,
                         float* __restrict__ fs, float* __restrict__ fd) {
    const int row = blockIdx.x * blockDim.x + threadIdx.x;
    const float* w = Wh + (size_t)row * OUT_DIM;
    float a = 0.f, b = 0.f;
#pragma unroll
    for (int c = 0; c < OUT_DIM; ++c) {
        float v = w[c];
        a = fmaf(v, attn[c], a);
        b = fmaf(v, attn[OUT_DIM + c], b);
    }
    fs[row] = a;
    fd[row] = b;
}

// ---------------- Phase 2b: Fmax = max_j fd[j] -----------------------------
__global__ void fmax_kernel(const float* __restrict__ fd, float* __restrict__ fmaxp) {
    __shared__ float red[1024];
    const int t = threadIdx.x;
    float m = -3.0e38f;
    for (int i = t; i < NN; i += 1024) m = fmaxf(m, fd[i]);
    red[t] = m;
    __syncthreads();
    for (int s = 512; s > 0; s >>= 1) {
        if (t < s) red[t] = fmaxf(red[t], red[t + s]);
        __syncthreads();
    }
    if (t == 0) fmaxp[0] = red[0];
}

// ---- Tensor Data Mover: 1-D 32KB copy fd(global) -> fdL(LDS) --------------
__device__ __forceinline__ void tdm_load_fd(const float* __restrict__ gsrc,
                                            float* lds_dst) {
    const unsigned long long ga = (unsigned long long)(uintptr_t)gsrc;
    const unsigned int ldsa = (unsigned int)(uintptr_t)lds_dst; // low 32b = LDS offset
    u32x4 g0;
    g0[0] = 1u;                                          // count=1, user D#
    g0[1] = ldsa;                                        // lds_addr
    g0[2] = (unsigned int)(ga & 0xFFFFFFFFu);            // global_addr[31:0]
    g0[3] = (unsigned int)((ga >> 32) & 0x01FFFFFFu) | (2u << 30); // [56:32]|type=2
    i32x8 g1;
    g1[0] = (int)(2u << 16);             // data_size = 4 bytes
    g1[1] = (int)((unsigned)NN << 16);   // tensor_dim0[15:0] = 8192
    g1[2] = (int)(1u << 16);             // tensor_dim0 hi=0 | tensor_dim1 = 1
    g1[3] = (int)((unsigned)NN << 16);   // tensor_dim1 hi=0 | tile_dim0 = 8192
    g1[4] = 1;                           // tile_dim1 = 1, tile_dim2 = 0
    g1[5] = NN;                          // tensor_dim0_stride = 8192
    g1[6] = 0;
    g1[7] = 0;
    i32x4 z4 = {};
#if defined(__clang_major__) && (__clang_major__ >= 23)
    i32x8 z8 = {};
    __builtin_amdgcn_tensor_load_to_lds(g0, g1, z4, z4, z8, 0);
#else
    __builtin_amdgcn_tensor_load_to_lds(g0, g1, z4, z4, 0);
#endif
}

// ---------------- Phase 3: fused mask + softmax + P@Wh ---------------------
// One workgroup (8 wave32) per 16-row block; wave w owns chunks {w, w+8, ...}.
__global__ __launch_bounds__(256) void attn_kernel(
        const int* __restrict__ A, const float* __restrict__ fs_g,
        const float* __restrict__ fmax_p, const float* __restrict__ fd_g,
        const _Float16* __restrict__ WhT, float* __restrict__ out) {
    __shared__ float fdL[NN];               // 32 KB: fd * log2(e)
    __shared__ float accL[OUT_DIM * 16];    // [col][M] cross-wave accumulator
    __shared__ float denomL[16];

    const int tid = threadIdx.x;
    if (tid < 32) {                          // one TDM issue (per-wave op)
        tdm_load_fd(fd_g, fdL);
        __builtin_amdgcn_s_wait_tensorcnt(0);
    }
    for (int i = tid; i < OUT_DIM * 16; i += 256) accL[i] = 0.f;
    if (tid < 16) denomL[tid] = 0.f;
    __syncthreads();
    for (int i = tid; i < NN; i += 256) fdL[i] *= LOG2E;   // log2 domain
    __syncthreads();

    const int lane    = tid & 31;
    const int wave    = tid >> 5;
    const int rowbase = blockIdx.x * 16;
    const int r       = lane & 15;   // tile row (A) / tile col (B,C,D)
    const int hi      = lane >> 4;
    const int kbase   = hi * 16;     // A-frag: lanes 16-31 own K+16
    const float Fmax  = fmax_p[0];
    const float fsr   = fs_g[rowbase + r];
    const float m2    = fmaxf(fsr + Fmax, 0.f) * LOG2E;  // relu(fs+Fmax)*log2e
    const float cc    = fsr * LOG2E - m2;
    const float negm2 = -m2;
    const int* Arow   = A + (size_t)(rowbase + r) * NN;

    v8f acc0 = {}, acc1 = {}, acc2 = {}, acc3 = {}, acc4 = {};

    for (int chunk = wave; chunk < NN / 32; chunk += 8) {
        const int j0 = chunk * 32;
        __builtin_prefetch(Arow + j0 + 8 * 32, 0, 0);   // this wave's next chunk

        // ---- A-fragment: 16x32 probability tile (contiguous K: k = 16*hi + e) ----
        int av[16];
        *(int4*)&av[0]  = *(const int4*)(Arow + j0 + kbase);
        *(int4*)&av[4]  = *(const int4*)(Arow + j0 + kbase + 4);
        *(int4*)&av[8]  = *(const int4*)(Arow + j0 + kbase + 8);
        *(int4*)&av[12] = *(const int4*)(Arow + j0 + kbase + 12);
        float fdv[16];
        *(float4*)&fdv[0]  = *(const float4*)&fdL[j0 + kbase];
        *(float4*)&fdv[4]  = *(const float4*)&fdL[j0 + kbase + 4];
        *(float4*)&fdv[8]  = *(const float4*)&fdL[j0 + kbase + 8];
        *(float4*)&fdv[12] = *(const float4*)&fdL[j0 + kbase + 12];

        v16h a;
#pragma unroll
        for (int e = 0; e < 16; ++e) {
            float arg = fmaxf(cc + fdv[e], negm2);      // log2 of relu'd score - m
            arg = (av[e] > 0) ? arg : -1.0e9f;          // branchless mask
            a[e] = (_Float16)__builtin_amdgcn_exp2f(arg);
        }

        // ---- B-fragments: WhT K-major, lane's 16 K-values contiguous ----
        const int jb = j0 + hi * 16;
        const _Float16* bp = WhT + (size_t)r * NN + jb;  // col = t*16 + r
        v16h b0 = *(const v16h*)(bp + (size_t)(0 * 16) * NN);
        v16h b1 = *(const v16h*)(bp + (size_t)(1 * 16) * NN);
        v16h b2 = *(const v16h*)(bp + (size_t)(2 * 16) * NN);
        v16h b3 = *(const v16h*)(bp + (size_t)(3 * 16) * NN);
        v16h b4 = *(const v16h*)(bp + (size_t)(4 * 16) * NN);  // ones column

        acc0 = __builtin_amdgcn_wmma_f32_16x16x32_f16(false, a, false, b0, (short)0, acc0, false, false);
        acc1 = __builtin_amdgcn_wmma_f32_16x16x32_f16(false, a, false, b1, (short)0, acc1, false, false);
        acc2 = __builtin_amdgcn_wmma_f32_16x16x32_f16(false, a, false, b2, (short)0, acc2, false, false);
        acc3 = __builtin_amdgcn_wmma_f32_16x16x32_f16(false, a, false, b3, (short)0, acc3, false, false);
        acc4 = __builtin_amdgcn_wmma_f32_16x16x32_f16(false, a, false, b4, (short)0, acc4, false, false);
    }

    // ---- combine waves / lane-halves via LDS float atomics ----
    if (r == 0) {                        // C layout: N=0 lives in lanes 0 and 16
#pragma unroll
        for (int v = 0; v < 8; ++v)
            atomicAdd(&denomL[v + hi * 8], acc4[v]);
    }
#pragma unroll
    for (int v = 0; v < 8; ++v) {
        const int M = v + hi * 8;        // C/D layout: M = vgpr + 8*hi
        atomicAdd(&accL[(0 * 16 + r) * 16 + M], acc0[v]);
        atomicAdd(&accL[(1 * 16 + r) * 16 + M], acc1[v]);
        atomicAdd(&accL[(2 * 16 + r) * 16 + M], acc2[v]);
        atomicAdd(&accL[(3 * 16 + r) * 16 + M], acc3[v]);
    }
    __syncthreads();

    // ---- normalize + relu + store ----
    for (int idx = tid; idx < 16 * OUT_DIM; idx += 256) {
        const int M = idx & 15;
        const int col = idx >> 4;
        const float d = denomL[M];
        const float val = (d > 0.f) ? accL[col * 16 + M] / d : 0.f;
        out[(size_t)(rowbase + M) * OUT_DIM + col] = fmaxf(val, 0.f);
    }
}

extern "C" void kernel_launch(void* const* d_in, const int* in_sizes, int n_in,
                              void* d_out, int out_size, void* d_ws, size_t ws_size,
                              hipStream_t stream) {
    (void)in_sizes; (void)n_in; (void)out_size; (void)ws_size;
    const float* X    = (const float*)d_in[0];
    const float* W    = (const float*)d_in[1];
    const float* attn = (const float*)d_in[2];
    const int*   A    = (const int*)d_in[3];
    float* out = (float*)d_out;

    char* ws = (char*)d_ws;
    float*    Wh    = (float*)ws;                                   // 2 MB
    _Float16* WhT   = (_Float16*)(ws + (size_t)NN * OUT_DIM * 4);   // 80 rows: 1.25 MB
    float*    fs    = (float*)(ws + 3584u * 1024u);                 // 32 KB
    float*    fd    = fs + NN;                                      // 32 KB
    float*    fmaxp = fd + NN;                                      // 4 B

    wh_kernel    <<<NN / 4,    256,  0, stream>>>(X, W, Wh, WhT);
    whtpad_kernel<<<16*NN/256, 256,  0, stream>>>(WhT);
    f_kernel     <<<NN / 256,  256,  0, stream>>>(Wh, attn, fs, fd);
    fmax_kernel  <<<1,         1024, 0, stream>>>(fd, fmaxp);
    attn_kernel  <<<NN / 16,   256,  0, stream>>>(A, fs, fmaxp, fd, WhT, out);
}
